// cf_gcn_17437567222109
// MI455X (gfx1250) — matrix operand order repaired and checked by reading, not compiled, not run on hardware
//
#include <hip/hip_runtime.h>
#include <hip/hip_bf16.h>
#include <math.h>
#include <stdint.h>

// ---------------- dims ----------------
#define BB    128
#define TT    32
#define VV    30000
#define VPAD  30080          // 235 * 128
#define WD    512
#define MF    64
#define EWD   200
#define HD    512
#define DOCN  500
#define GRUIN 640            // WD + 2*MF
#define G3    1536           // 3*HD

typedef __attribute__((ext_vector_type(16))) __bf16 v16bf;
typedef __attribute__((ext_vector_type(8)))  float  v8f;

static __device__ __forceinline__ unsigned short f2bfu(float f) {
    union { float f; unsigned u; } v; v.f = f;
    unsigned r = (v.u + 0x7FFFu + ((v.u >> 16) & 1u)) >> 16;
    return (unsigned short)r;
}
static __device__ __forceinline__ __bf16 us2bf(unsigned short s) {
    union { unsigned short u; __bf16 b; } v; v.u = s; return v.b;
}

// ---------------- prep: bf16 weight conversion, padding, h0 zero ----------------
__global__ __launch_bounds__(256) void k_prep(
    const float* __restrict__ Wih, const float* __restrict__ Whh,
    const float* __restrict__ outW, const float* __restrict__ outb,
    unsigned short* __restrict__ WihBf, unsigned short* __restrict__ WhhBf,
    unsigned short* __restrict__ outWBf, float* __restrict__ outbPad,
    float* __restrict__ h0)
{
    const size_t N = (size_t)VPAD * WD;  // largest array
    size_t stride = (size_t)gridDim.x * blockDim.x;
    for (size_t i = (size_t)blockIdx.x * blockDim.x + threadIdx.x; i < N; i += stride) {
        if (i < (size_t)G3 * GRUIN) WihBf[i] = f2bfu(Wih[i]);
        if (i < (size_t)G3 * HD)    WhhBf[i] = f2bfu(Whh[i]);
        {   // padded out_W (zeros past VV)
            size_t row = i >> 9;  // /512
            outWBf[i] = (row < VV) ? f2bfu(outW[i]) : (unsigned short)0;
        }
        if (i < VPAD) outbPad[i] = (i < VV) ? outb[i] : -1e30f;
        if (i < (size_t)BB * HD) h0[i] = 0.0f;
    }
}

// ---------------- encoder + rating decoder ----------------
__global__ __launch_bounds__(256) void k_encoder(
    const int* __restrict__ user, const int* __restrict__ item,
    const float* __restrict__ label,
    const int* __restrict__ user_doc, const int* __restrict__ item_doc,
    const float* __restrict__ gamma_user, const float* __restrict__ gamma_item,
    const float* __restrict__ theta_user, const float* __restrict__ theta_item,
    const float* __restrict__ doc_emb,
    const float* __restrict__ W_tu, const float* __restrict__ W_ti,
    const float* __restrict__ pred_W, const float* __restrict__ pred_b,
    const float* __restrict__ b_users, const float* __restrict__ b_items,
    float* __restrict__ rating_loss, float* __restrict__ gu_out, float* __restrict__ gi_out)
{
    const int b = blockIdx.x, tid = threadIdx.x;
    __shared__ float ud[EWD], idc[EWD], smlp[3 * MF], sred[256];
    const int u  = user[b];
    const int it = item[b];
    if (tid < EWD) {
        float su = 0.f, si = 0.f;
        for (int d = 0; d < DOCN; ++d) {
            su += doc_emb[(size_t)user_doc[b * DOCN + d] * EWD + tid];
            si += doc_emb[(size_t)item_doc[b * DOCN + d] * EWD + tid];
        }
        ud[tid]  = su * (1.0f / DOCN);
        idc[tid] = si * (1.0f / DOCN);
    }
    __syncthreads();
    if (tid < MF) {
        float guv = gamma_user[(size_t)u * MF + tid];
        float giv = gamma_item[(size_t)it * MF + tid];
        gu_out[b * MF + tid] = guv;
        gi_out[b * MF + tid] = giv;
        smlp[tid] = guv * giv;
        float d1 = 0.f, d2 = 0.f;
        for (int e = 0; e < EWD; ++e) {
            d1 += ud[e]  * W_tu[tid * EWD + e];
            d2 += idc[e] * W_ti[tid * EWD + e];
        }
        smlp[MF + tid]     = theta_user[(size_t)u * MF + tid] * d1;
        smlp[2 * MF + tid] = theta_item[(size_t)it * MF + tid] * d2;
    }
    __syncthreads();
    float p = (tid < 3 * MF) ? smlp[tid] * pred_W[tid] : 0.f;
    sred[tid] = p;
    __syncthreads();
    for (int o = 128; o > 0; o >>= 1) {
        if (tid < o) sred[tid] += sred[tid + o];
        __syncthreads();
    }
    if (tid == 0) {
        float pred = sred[0] + pred_b[0] + 3.5f + b_users[u] + b_items[it];
        float diff = pred - label[b];
        rating_loss[b] = diff * diff;
    }
}

// ---------------- build x_seq bf16: [T*B, 640] ----------------
__global__ __launch_bounds__(256) void k_build_x(
    const int* __restrict__ review_input, const float* __restrict__ word_emb,
    const float* __restrict__ gu, const float* __restrict__ gi,
    unsigned short* __restrict__ xbf)
{
    const int row = blockIdx.x;         // t*B + b
    const int b   = row & (BB - 1);
    const int w   = review_input[row];
    for (int c = threadIdx.x; c < GRUIN; c += blockDim.x) {
        float v;
        if (c < WD)            v = word_emb[(size_t)w * WD + c];
        else if (c < WD + MF)  v = gu[b * MF + (c - WD)];
        else                   v = gi[b * MF + (c - WD - MF)];
        xbf[(size_t)row * GRUIN + c] = f2bfu(v);
    }
}

// ---------------- gi_all = X @ W_ih.T + b_ih  : [4096,640]x[640->1536] ----------------
__global__ __launch_bounds__(256) void k_gemm_gi(
    const unsigned short* __restrict__ xbf, const unsigned short* __restrict__ Wih,
    const float* __restrict__ b_ih, float* __restrict__ gi_all)
{
    __shared__ unsigned short sA[16 * GRUIN];   // 20 KB
    const int rt = blockIdx.x;                  // 0..255 row tiles
    const int cg = blockIdx.y;                  // 0..11  (12 * 8 waves * 16 = 1536)
    const int tid = threadIdx.x;
    const int row0 = rt * 16;
    for (int i = tid; i < 16 * GRUIN; i += 256) {
        int m = i / GRUIN, k = i % GRUIN;
        sA[i] = xbf[(size_t)(row0 + m) * GRUIN + k];
    }
    __syncthreads();
    const int wave = tid >> 5, lane = tid & 31;
    const int lm = lane & 15, lh = lane >> 4;
    const int n0 = cg * 128 + wave * 16;
    v8f acc = {};
    for (int kk = 0; kk < GRUIN / 32; ++kk) {
        const int kb = kk * 32;
        v16bf a, bmat;
#pragma unroll
        for (int i = 0; i < 8; ++i) {
            a[i]     = us2bf(sA[lm * GRUIN + kb + lh * 16 + i]);
            a[i + 8] = us2bf(sA[lm * GRUIN + kb + lh * 16 + 8 + i]);
        }
        const unsigned short* brow = Wih + (size_t)(n0 + lm) * GRUIN + kb + lh * 16;
#pragma unroll
        for (int i = 0; i < 16; ++i) bmat[i] = us2bf(brow[i]);
        acc = __builtin_amdgcn_wmma_f32_16x16x32_bf16(false, a, false, bmat,
                                                      (short)0, acc, false, false);
    }
    const int col = n0 + lm;
    const float bias = b_ih[col];
#pragma unroll
    for (int r = 0; r < 8; ++r) {
        int m = r + 8 * lh;
        gi_all[(size_t)(row0 + m) * G3 + col] = acc[r] + bias;
    }
}

// ---------------- one GRU step: gh = h@W_hh.T + b_hh, gate fusion ----------------
__global__ __launch_bounds__(128) void k_gru_step(
    const float* __restrict__ h_in, const float* __restrict__ gi_t,
    const unsigned short* __restrict__ Whh, const float* __restrict__ b_hh,
    float* __restrict__ h_out, unsigned short* __restrict__ out_bf)
{
    __shared__ unsigned short sA[16 * HD];      // 16 KB (h tile, bf16)
    const int rt = blockIdx.x;                  // 0..7 (128 rows / 16)
    const int cgrp = blockIdx.y;                // 0..7 (512 cols / (4 waves * 16))
    const int tid = threadIdx.x;
    const int row0 = rt * 16;
    for (int i = tid; i < 16 * HD; i += 128) {
        int m = i / HD, k = i % HD;
        sA[i] = f2bfu(h_in[(size_t)(row0 + m) * HD + k]);
    }
    __syncthreads();
    const int wave = tid >> 5, lane = tid & 31;
    const int lm = lane & 15, lh = lane >> 4;
    const int j0 = cgrp * 64 + wave * 16;
    v8f acc0 = {}, acc1 = {}, acc2 = {};
    for (int kk = 0; kk < HD / 32; ++kk) {
        const int kb = kk * 32;
        v16bf a, b0, b1, b2;
#pragma unroll
        for (int i = 0; i < 8; ++i) {
            a[i]     = us2bf(sA[lm * HD + kb + lh * 16 + i]);
            a[i + 8] = us2bf(sA[lm * HD + kb + lh * 16 + 8 + i]);
        }
        const unsigned short* r0p = Whh + (size_t)(j0 + lm) * HD + kb + lh * 16;
        const unsigned short* r1p = Whh + (size_t)(j0 + HD + lm) * HD + kb + lh * 16;
        const unsigned short* r2p = Whh + (size_t)(j0 + 2 * HD + lm) * HD + kb + lh * 16;
#pragma unroll
        for (int i = 0; i < 16; ++i) { b0[i] = us2bf(r0p[i]); b1[i] = us2bf(r1p[i]); b2[i] = us2bf(r2p[i]); }
        acc0 = __builtin_amdgcn_wmma_f32_16x16x32_bf16(false, a, false, b0, (short)0, acc0, false, false);
        acc1 = __builtin_amdgcn_wmma_f32_16x16x32_bf16(false, a, false, b1, (short)0, acc1, false, false);
        acc2 = __builtin_amdgcn_wmma_f32_16x16x32_bf16(false, a, false, b2, (short)0, acc2, false, false);
    }
    const int col = j0 + lm;
    const float bh_r = b_hh[col], bh_z = b_hh[col + HD], bh_n = b_hh[col + 2 * HD];
#pragma unroll
    for (int r = 0; r < 8; ++r) {
        int m = r + 8 * lh;
        int grow = row0 + m;
        float i_r = gi_t[(size_t)grow * G3 + col];
        float i_z = gi_t[(size_t)grow * G3 + col + HD];
        float i_n = gi_t[(size_t)grow * G3 + col + 2 * HD];
        float h_r = acc0[r] + bh_r;
        float h_z = acc1[r] + bh_z;
        float h_n = acc2[r] + bh_n;
        float rg = 1.0f / (1.0f + __expf(-(i_r + h_r)));
        float zg = 1.0f / (1.0f + __expf(-(i_z + h_z)));
        float ng = tanhf(i_n + rg * h_n);
        float hp = h_in[(size_t)grow * HD + col];
        float hn = (1.0f - zg) * ng + zg * hp;
        h_out[(size_t)grow * HD + col] = hn;
        out_bf[(size_t)grow * HD + col] = f2bfu(hn);
    }
}

// -------- fused logits GEMM + online log-softmax + NLL (32 rows/block, B reuse x2) --------
__global__ __launch_bounds__(256) void k_logits_nll(
    const unsigned short* __restrict__ out_bf, const unsigned short* __restrict__ WoutBf,
    const float* __restrict__ outbPad, const int* __restrict__ tgt,
    float* __restrict__ d_loss)
{
    __shared__ unsigned short sA[32 * HD];          // 32 KB (A tile, 2 row-tiles)
    __shared__ float sPM[8 * 32 * 16];              // 16 KB
    __shared__ float sPS[8 * 32 * 16];              // 16 KB
    __shared__ float sTL[32];
    __shared__ int   sTG[32];
    const int rt = blockIdx.x;                      // 0..127 (32 rows each)
    const int tid = threadIdx.x;
    const int row0 = rt * 32;

    // Async DMA of the contiguous 32x512 bf16 A tile into LDS (ASYNCcnt path).
    {
        const unsigned short* gsrc = out_bf + (size_t)row0 * HD;
        for (int i = tid; i < (32 * HD) / 8; i += 256) {     // 16-byte chunks
            unsigned lds = (unsigned)(uintptr_t)(&sA[i * 8]);
            uint64_t ga = (uint64_t)(uintptr_t)(gsrc + i * 8);
            asm volatile("global_load_async_to_lds_b128 %0, %1, off"
                         :: "v"(lds), "v"(ga) : "memory");
        }
        asm volatile("s_wait_asynccnt 0" ::: "memory");
    }
    if (tid < 32) { sTG[tid] = tgt[row0 + tid]; sTL[tid] = 0.0f; }
    __syncthreads();

    const int wave = tid >> 5, lane = tid & 31;
    const int lm = lane & 15, lh = lane >> 4;
    float runM[16], runS[16];
#pragma unroll
    for (int r = 0; r < 16; ++r) { runM[r] = -1e30f; runS[r] = 0.0f; }

    for (int chunk = 0; chunk < VPAD / 128; ++chunk) {
        const int n0 = chunk * 128 + wave * 16;
        // prefetch next chunk's B row for this lane (L2-resident stream)
        if (chunk + 1 < VPAD / 128)
            __builtin_prefetch(WoutBf + (size_t)(n0 + 128 + lm) * HD, 0, 1);
        v8f acc0 = {}, acc1 = {};
        for (int kk = 0; kk < HD / 32; ++kk) {
            const int kb = kk * 32;
            v16bf a0, a1, bmat;
#pragma unroll
            for (int i = 0; i < 8; ++i) {
                a0[i]     = us2bf(sA[lm * HD + kb + lh * 16 + i]);
                a0[i + 8] = us2bf(sA[lm * HD + kb + lh * 16 + 8 + i]);
                a1[i]     = us2bf(sA[(16 + lm) * HD + kb + lh * 16 + i]);
                a1[i + 8] = us2bf(sA[(16 + lm) * HD + kb + lh * 16 + 8 + i]);
            }
            const unsigned short* brow = WoutBf + (size_t)(n0 + lm) * HD + kb + lh * 16;
#pragma unroll
            for (int i = 0; i < 16; ++i) bmat[i] = us2bf(brow[i]);
            acc0 = __builtin_amdgcn_wmma_f32_16x16x32_bf16(false, a0, false, bmat,
                                                           (short)0, acc0, false, false);
            acc1 = __builtin_amdgcn_wmma_f32_16x16x32_bf16(false, a1, false, bmat,
                                                           (short)0, acc1, false, false);
        }
        const int col = n0 + lm;
        const float bias = outbPad[col];
#pragma unroll
        for (int r = 0; r < 8; ++r) {
            int m0 = r + 8 * lh;
            float v0 = acc0[r] + bias;
            float v1 = acc1[r] + bias;
            if (col == sTG[m0])      sTL[m0]      = v0;  // unique (row,col): race-free
            if (col == sTG[m0 + 16]) sTL[m0 + 16] = v1;
            float nm0 = fmaxf(runM[r], v0);
            runS[r] = runS[r] * __expf(runM[r] - nm0) + __expf(v0 - nm0);
            runM[r] = nm0;
            float nm1 = fmaxf(runM[r + 8], v1);
            runS[r + 8] = runS[r + 8] * __expf(runM[r + 8] - nm1) + __expf(v1 - nm1);
            runM[r + 8] = nm1;
        }
    }
#pragma unroll
    for (int r = 0; r < 8; ++r) {
        int base = (wave * 32 + lane) << 4;
        sPM[base + r]     = runM[r];      // tile0 rows
        sPS[base + r]     = runS[r];
        sPM[base + 8 + r] = runM[r + 8];  // tile1 rows
        sPS[base + 8 + r] = runS[r + 8];
    }
    __syncthreads();
    if (tid < 32) {
        const int m = tid;                 // local row 0..31
        const int tile = m >> 4, mm = m & 15;
        const int half = mm >> 3, r = mm & 7;
        float M = -1e30f, S = 0.0f;
        for (int w = 0; w < 8; ++w)
            for (int c = 0; c < 16; ++c) {
                int lane2 = half * 16 + c;
                int idx = ((w * 32 + lane2) << 4) + tile * 8 + r;
                float pm = sPM[idx], ps = sPS[idx];
                float nm = fmaxf(M, pm);
                S = S * __expf(M - nm) + ps * __expf(pm - nm);
                M = nm;
            }
        float nll = -(sTL[m] - M - __logf(S));
        int rowg = row0 + m;
        d_loss[rowg] = (sTG[m] != 0) ? nll : 0.0f;
    }
}

// ---------------- final reduction: review_obj, rating_obj, obj ----------------
__global__ __launch_bounds__(256) void k_final(float* __restrict__ d_out,
                                               const int* __restrict__ tgt)
{
    __shared__ float s1[256], s2[256];
    const int tid = threadIdx.x;
    float sn = 0.f, sm = 0.f;
    for (int i = tid; i < TT * BB; i += 256) {
        sn += d_out[i];
        sm += (tgt[i] != 0) ? 1.0f : 0.0f;
    }
    s1[tid] = sn; s2[tid] = sm;
    __syncthreads();
    for (int o = 128; o > 0; o >>= 1) {
        if (tid < o) { s1[tid] += s1[tid + o]; s2[tid] += s2[tid + o]; }
        __syncthreads();
    }
    float nll_sum = s1[0], mask_sum = s2[0];
    __syncthreads();
    float rr = 0.f;
    for (int i = tid; i < BB; i += 256) rr += d_out[TT * BB + i];
    s1[tid] = rr;
    __syncthreads();
    for (int o = 128; o > 0; o >>= 1) {
        if (tid < o) s1[tid] += s1[tid + o];
        __syncthreads();
    }
    if (tid == 0) {
        float review_obj = nll_sum / fmaxf(mask_sum, 1.0f);
        d_out[TT * BB + BB] = s1[0] + 1e-10f * review_obj;
    }
}

// ---------------- host launch ----------------
static inline size_t align256(size_t x) { return (x + 255u) & ~(size_t)255u; }

extern "C" void kernel_launch(void* const* d_in, const int* in_sizes, int n_in,
                              void* d_out, int out_size, void* d_ws, size_t ws_size,
                              hipStream_t stream) {
    (void)in_sizes; (void)n_in; (void)out_size; (void)ws_size;
    const int*   user       = (const int*)  d_in[0];
    const int*   item       = (const int*)  d_in[1];
    const float* label      = (const float*)d_in[2];
    const int*   review_in  = (const int*)  d_in[3];
    const int*   review_tgt = (const int*)  d_in[4];
    const int*   user_doc   = (const int*)  d_in[5];
    const int*   item_doc   = (const int*)  d_in[6];
    const float* gamma_user = (const float*)d_in[7];
    const float* gamma_item = (const float*)d_in[8];
    const float* theta_user = (const float*)d_in[9];
    const float* theta_item = (const float*)d_in[10];
    const float* doc_emb    = (const float*)d_in[11];
    const float* word_emb   = (const float*)d_in[12];
    const float* W_tu       = (const float*)d_in[13];
    const float* W_ti       = (const float*)d_in[14];
    const float* pred_W     = (const float*)d_in[15];
    const float* pred_b     = (const float*)d_in[16];
    const float* b_users    = (const float*)d_in[17];
    const float* b_items    = (const float*)d_in[18];
    const float* W_ih       = (const float*)d_in[19];
    const float* W_hh       = (const float*)d_in[20];
    const float* b_ih       = (const float*)d_in[21];
    const float* b_hh       = (const float*)d_in[22];
    const float* out_W      = (const float*)d_in[23];
    const float* out_b      = (const float*)d_in[24];

    float* out = (float*)d_out;   // [4096 nll | 128 rating | 1 obj]

    // workspace carve-up
    char* ws = (char*)d_ws;
    size_t off = 0;
    unsigned short* xbf     = (unsigned short*)(ws + off); off += align256((size_t)TT * BB * GRUIN * 2);
    unsigned short* WihBf   = (unsigned short*)(ws + off); off += align256((size_t)G3 * GRUIN * 2);
    unsigned short* WhhBf   = (unsigned short*)(ws + off); off += align256((size_t)G3 * HD * 2);
    unsigned short* outWBf  = (unsigned short*)(ws + off); off += align256((size_t)VPAD * HD * 2);
    float*          outbPad = (float*)         (ws + off); off += align256((size_t)VPAD * 4);
    float*          gi_all  = (float*)         (ws + off); off += align256((size_t)TT * BB * G3 * 4);
    float*          h0      = (float*)         (ws + off); off += align256((size_t)BB * HD * 4);
    float*          h1      = (float*)         (ws + off); off += align256((size_t)BB * HD * 4);
    unsigned short* outSeq  = (unsigned short*)(ws + off); off += align256((size_t)TT * BB * HD * 2);
    float*          guBuf   = (float*)         (ws + off); off += align256((size_t)BB * MF * 4);
    float*          giBuf   = (float*)         (ws + off); off += align256((size_t)BB * MF * 4);

    // 1. weight conversion + padding + h0 zero
    k_prep<<<8192, 256, 0, stream>>>(W_ih, W_hh, out_W, out_b,
                                     WihBf, WhhBf, outWBf, outbPad, h0);
    // 2. encoder + rating loss (writes out[4096..4223], gu/gi buffers)
    k_encoder<<<BB, 256, 0, stream>>>(user, item, label, user_doc, item_doc,
                                      gamma_user, gamma_item, theta_user, theta_item,
                                      doc_emb, W_tu, W_ti, pred_W, pred_b,
                                      b_users, b_items, out + TT * BB, guBuf, giBuf);
    // 3. x_seq bf16
    k_build_x<<<TT * BB, 256, 0, stream>>>(review_in, word_emb, guBuf, giBuf, xbf);
    // 4. input-side GRU GEMM (all T at once, WMMA)
    k_gemm_gi<<<dim3(TT * BB / 16, G3 / 128), 256, 0, stream>>>(xbf, WihBf, b_ih, gi_all);
    // 5. recurrent steps (WMMA + fused gates), h ping-pong
    for (int t = 0; t < TT; ++t) {
        const float* hin  = (t & 1) ? h1 : h0;
        float*       hout = (t & 1) ? h0 : h1;
        k_gru_step<<<dim3(BB / 16, HD / 64), 128, 0, stream>>>(
            hin, gi_all + (size_t)t * BB * G3, WhhBf, b_hh,
            hout, outSeq + (size_t)t * BB * HD);
    }
    // 6. fused logits GEMM + online log-softmax + NLL (writes out[0..4095])
    k_logits_nll<<<TT * BB / 32, 256, 0, stream>>>(outSeq, outWBf, outbPad, review_tgt, out);
    // 7. objectives (writes out[4224])
    k_final<<<1, 256, 0, stream>>>(out, review_tgt);
}
